// MACNet_46926812677062
// MI455X (gfx1250) — matrix operand order repaired and checked by reading, not compiled, not run on hardware
//
#include <hip/hip_runtime.h>
#include <math.h>

typedef __bf16 bf16_t;
typedef __attribute__((ext_vector_type(16))) __bf16        v16bf;
typedef __attribute__((ext_vector_type(8)))  float         v8f;
typedef __attribute__((ext_vector_type(4)))  unsigned int  v4u;
typedef __attribute__((ext_vector_type(8)))  int           v8i;
typedef __attribute__((ext_vector_type(4)))  int           v4i;

// ---- problem dims ----
#define NBATCH 128
#define SEQ    200
#define HIDN   1024
#define NH     4
#define DKD    256
#define NFREQ  101               // MAXSEQ/2 + 1
#define FROWS  202               // [Re(0..100); Im(0..100)]
#define FPAD   208               // padded row stride for freq buffers (16B align)
#define BHN    (NBATCH * NH)     // 512

// ---- GEMM tiling ----
#define BM 128
#define BN 128
#define BK 32
#define NTHR 256                 // 8 waves
#define TLDS 40                  // LDS row stride: 32 data + 8 pad (bf16 elements)

enum { EPI_F32 = 0, EPI_BF16 = 1, EPI_BIAS_RELU_BF16 = 2, EPI_BIAS_F32 = 3,
       EPI_T_BF16 = 4 };

#if __has_builtin(__builtin_amdgcn_tensor_load_to_lds) && \
    __has_builtin(__builtin_amdgcn_s_wait_tensorcnt)
#define USE_TDM 1
#else
#define USE_TDM 0
#endif

// ---------------------------------------------------------------------------
// TDM load of one (tileRows x BK) bf16 tile from global into LDS with
// hardware row padding (16 data DWORDs + 4 pad DWORDs == TLDS=40 elements).
// tensor dims = REMAINING extents from the tile origin, so hardware zero-fills
// out-of-bounds reads (no staging guards needed).
// ---------------------------------------------------------------------------
#if USE_TDM
__device__ __forceinline__ void tdm_load_tile(const bf16_t* g, unsigned ldsOff,
                                              int rowsRem, int colsRem, int ld,
                                              int tileRows)
{
  const unsigned long long ga = (unsigned long long)(size_t)(const void*)g;
  v4u g0;
  g0[0] = 1u;                                       // count=1, user mode
  g0[1] = ldsOff;                                   // lds_addr
  g0[2] = (unsigned)(ga & 0xffffffffull);           // global_addr[31:0]
  g0[3] = (unsigned)((ga >> 32) & 0x01ffffffull)    // global_addr[56:32]
          | (2u << 30);                             // type = 2 ("image")
  const unsigned d0 = (unsigned)colsRem;            // tensor_dim0 (k)
  const unsigned d1 = (unsigned)rowsRem;            // tensor_dim1 (rows)
  v8i g1;
  g1[0] = (int)((1u << 16)      // data_size = 2 bytes
              | (1u << 20)      // pad_enable
              | (3u << 22)      // pad_interval: 16 DWORDs (= one 32-elem row)
              | (3u << 25));    // pad_amount:   4 DWORDs (= 8 elements)
  g1[1] = (int)((d0 & 0xffffu) << 16);                          // dim0[15:0]
  g1[2] = (int)(((d0 >> 16) & 0xffffu) | ((d1 & 0xffffu) << 16));
  g1[3] = (int)(((d1 >> 16) & 0xffffu) | ((unsigned)BK << 16)); // tile_dim0
  g1[4] = tileRows;                                              // tile_dim1
  g1[5] = (int)(unsigned)ld;                                     // dim0 stride
  g1[6] = 0; g1[7] = 0;
  const v4i gz = {0, 0, 0, 0};
#if defined(__clang_major__) && __clang_major__ >= 23
  const v8i gz8 = {0, 0, 0, 0, 0, 0, 0, 0};
  __builtin_amdgcn_tensor_load_to_lds(g0, g1, gz, gz, gz8, 0);
#else
  __builtin_amdgcn_tensor_load_to_lds(g0, g1, gz, gz, 0);
#endif
}
#else
// Fallback: branch-free two-phase staging (all loads issued, one wait, stores)
__device__ __forceinline__ void stage_tile(bf16_t* dst, const bf16_t* g,
                                           int rowsRem, int colsRem, int ld,
                                           int tid)
{
  bf16_t vals[16];
#pragma unroll
  for (int it = 0; it < 16; ++it) {
    const int idx = tid + it * NTHR;
    const int k = idx & 31, r = idx >> 5;
    const int rc = r < rowsRem ? r : rowsRem - 1;
    const int kc = k < colsRem ? k : colsRem - 1;
    const bf16_t t = g[(size_t)rc * ld + kc];
    vals[it] = (r < rowsRem && k < colsRem) ? t : (bf16_t)0.0f;
  }
#pragma unroll
  for (int it = 0; it < 16; ++it) {
    const int idx = tid + it * NTHR;
    dst[(idx >> 5) * TLDS + (idx & 31)] = vals[it];
  }
}
#endif

// ---------------------------------------------------------------------------
// Batched bf16 WMMA GEMM:  C(MxN) = A(MxK) * B(KxN), B given as BT[N][K].
// Both operands k-contiguous bf16 -> both tiles are plain 2-D TDM loads.
// 256 threads = 8 waves in a 4x2 grid; each wave owns a 32x64 quadrant of the
// 128x128 block tile => 8x v_wmma_f32_16x16x32_bf16 per 32-deep K step.
// Double-buffered: wave0 prefetches A tiles, wave1 prefetches B tiles.
// ---------------------------------------------------------------------------
template <int EPI>
__global__ __launch_bounds__(NTHR) void wmma_gemm(
    const bf16_t* __restrict__ A,  int lda, long long aO, long long aI,
    const bf16_t* __restrict__ BT, int ldb, long long bO, long long bI,
    void*         __restrict__ Cp, int ldc, long long cO, long long cI,
    const float* __restrict__ bias, int M, int N, int K, int innerDiv)
{
  __shared__ bf16_t As[2][BM * TLDS];
  __shared__ bf16_t Bs[2][BN * TLDS];

  const int z  = blockIdx.z;
  const int zo = z / innerDiv, zi = z - zo * innerDiv;
  const bf16_t* Ab = A  + (size_t)zo * aO + (size_t)zi * aI;
  const bf16_t* Bb = BT + (size_t)zo * bO + (size_t)zi * bI;
  float*  C32 = (float*)Cp  + (size_t)zo * cO + (size_t)zi * cI;
  bf16_t* C16 = (bf16_t*)Cp + (size_t)zo * cO + (size_t)zi * cI;
  (void)C32; (void)C16;

  const int m0   = blockIdx.y * BM;
  const int n0   = blockIdx.x * BN;
  const int tid  = threadIdx.x;
  const int lane = tid & 31;
  const int wave = tid >> 5;
  const int wm   = (wave >> 1) * 32;   // 0,32,64,96
  const int wn   = (wave & 1) * 64;    // 0,64

  v8f acc[2][4];
#pragma unroll
  for (int i = 0; i < 2; ++i)
#pragma unroll
    for (int j = 0; j < 4; ++j)
#pragma unroll
      for (int e = 0; e < 8; ++e) acc[i][j][e] = 0.0f;

  const int nk = (K + BK - 1) / BK;

#if USE_TDM
  // prologue: prefetch tile 0
  if (wave == 0)
    tdm_load_tile(Ab + (size_t)m0 * lda, (unsigned)(size_t)(const void*)&As[0][0],
                  M - m0, K, lda, BM);
  else if (wave == 1)
    tdm_load_tile(Bb + (size_t)n0 * ldb, (unsigned)(size_t)(const void*)&Bs[0][0],
                  N - n0, K, ldb, BN);
#endif

  for (int it = 0; it < nk; ++it) {
#if USE_TDM
    const int cur = it & 1;
    const bool more = (it + 1 < nk);
    if (more) {
      const int kn = (it + 1) * BK;
      if (wave == 0)
        tdm_load_tile(Ab + (size_t)m0 * lda + kn,
                      (unsigned)(size_t)(const void*)&As[cur ^ 1][0],
                      M - m0, K - kn, lda, BM);
      else if (wave == 1)
        tdm_load_tile(Bb + (size_t)n0 * ldb + kn,
                      (unsigned)(size_t)(const void*)&Bs[cur ^ 1][0],
                      N - n0, K - kn, ldb, BN);
    }
    if (wave < 2) {
      if (more) __builtin_amdgcn_s_wait_tensorcnt(1);
      else      __builtin_amdgcn_s_wait_tensorcnt(0);
    }
    __syncthreads();
#else
    const int cur = 0;
    const int kt = it * BK;
    stage_tile(&As[0][0], Ab + (size_t)m0 * lda + kt, M - m0, K - kt, lda, tid);
    stage_tile(&Bs[0][0], Bb + (size_t)n0 * ldb + kt, N - n0, K - kt, ldb, tid);
    __syncthreads();
#endif

    // fragment loads: contiguous k = 16*(lane/16) + e for both A and B (two b128 reads each)
    const int am  = lane & 15;
    const int akb = (lane >> 4) * 16;
    const int bkb = (lane >> 4) * 16;

    v16bf afr[2], bfr[4];
#pragma unroll
    for (int i = 0; i < 2; ++i) {
      const bf16_t* s = &As[cur][(wm + 16 * i + am) * TLDS];
#pragma unroll
      for (int e = 0; e < 8; ++e) {
        afr[i][e]     = s[akb + e];
        afr[i][8 + e] = s[8 + akb + e];
      }
    }
#pragma unroll
    for (int j = 0; j < 4; ++j) {
      const bf16_t* s = &Bs[cur][(wn + 16 * j + am) * TLDS + bkb];
#pragma unroll
      for (int e = 0; e < 16; ++e) bfr[j][e] = s[e];
    }
#pragma unroll
    for (int i = 0; i < 2; ++i)
#pragma unroll
      for (int j = 0; j < 4; ++j)
        acc[i][j] = __builtin_amdgcn_wmma_f32_16x16x32_bf16(
            false, afr[i], false, bfr[j], (short)0, acc[i][j], false, false);
    __syncthreads();
  }

  // ---- epilogues (C layout: VGPR r -> M = r + 8*(lane/16), N = lane%16) ----
  const int cn  = lane & 15;
  const int crs = (lane >> 4) * 8;
#pragma unroll
  for (int i = 0; i < 2; ++i) {
#pragma unroll
    for (int j = 0; j < 4; ++j) {
      const int gn  = n0 + wn + 16 * j + cn;
      const int gm0 = m0 + wm + 16 * i + crs;
      if constexpr (EPI == EPI_T_BF16) {
        // transposed store: CT[n][m], 8 contiguous m per lane -> one b128
        if (gn < N) {
          if (gm0 + 7 < M) {
            union { bf16_t h[8]; v4u v; } u;
#pragma unroll
            for (int r = 0; r < 8; ++r) u.h[r] = (bf16_t)acc[i][j][r];
            *(v4u*)&C16[(size_t)gn * ldc + gm0] = u.v;
          } else {
#pragma unroll
            for (int r = 0; r < 8; ++r)
              if (gm0 + r < M)
                C16[(size_t)gn * ldc + gm0 + r] = (bf16_t)acc[i][j][r];
          }
        }
      } else {
#pragma unroll
        for (int r = 0; r < 8; ++r) {
          const int gm = gm0 + r;
          if (gm < M && gn < N) {
            float v = acc[i][j][r];
            if constexpr (EPI == EPI_BIAS_RELU_BF16) {
              v += bias[gn]; v = fmaxf(v, 0.0f);
              C16[(size_t)gm * ldc + gn] = (bf16_t)v;
            } else if constexpr (EPI == EPI_BIAS_F32) {
              v += bias[gn];
              C32[(size_t)gm * ldc + gn] = v;
            } else if constexpr (EPI == EPI_BF16) {
              C16[(size_t)gm * ldc + gn] = (bf16_t)v;
            } else {
              C32[(size_t)gm * ldc + gn] = v;
            }
          }
        }
      }
    }
  }
}

// ---------------------------------------------------------------------------
// small helpers
// ---------------------------------------------------------------------------
__global__ void cvt_f32_bf16(const float* __restrict__ s, bf16_t* __restrict__ d,
                             long long n)
{
  const long long i = (long long)blockIdx.x * blockDim.x + threadIdx.x;
  if (i < n) d[i] = (bf16_t)s[i];
}

// d[c][r] = s[r][c]   (s is R x C)
__global__ void cvt_f32_bf16_t(const float* __restrict__ s, bf16_t* __restrict__ d,
                               int R, int C)
{
  const int i = blockIdx.x * blockDim.x + threadIdx.x;
  if (i >= R * C) return;
  const int r = i / C, c = i - r * C;
  d[(size_t)c * R + r] = (bf16_t)s[i];
}

// Ortho real-DFT matrices, generated directly in bf16.
//   F (FROWS x SEQ): rows 0..100 cos/√n ; rows 101..201 -sin/√n
//   G (SEQ x FROWS): w_f cos/√n and -w_f sin/√n (w=1 at f=0,n/2 else 2)
__global__ void init_dft(bf16_t* __restrict__ F, bf16_t* __restrict__ G)
{
  const int idx = blockIdx.x * blockDim.x + threadIdx.x;
  if (idx >= FROWS * SEQ) return;
  const float TWO_PI = 6.283185307179586f;
  const float inv = 0.07071067811865475f;   // 1/sqrt(200)
  {
    const int f = idx / SEQ, t = idx - f * SEQ;
    float v;
    if (f < NFREQ) v = cosf(TWO_PI * (float)(f * t) / (float)SEQ) * inv;
    else v = -sinf(TWO_PI * (float)((f - NFREQ) * t) / (float)SEQ) * inv;
    F[idx] = (bf16_t)v;
  }
  {
    const int t = idx / FROWS, f = idx - t * FROWS;
    float v;
    if (f < NFREQ) {
      const float w = (f == 0 || f == NFREQ - 1) ? 1.0f : 2.0f;
      v = w * cosf(TWO_PI * (float)(f * t) / (float)SEQ) * inv;
    } else {
      const int ff = f - NFREQ;
      const float w = (ff == 0 || ff == NFREQ - 1) ? 1.0f : 2.0f;
      v = -w * sinf(TWO_PI * (float)(ff * t) / (float)SEQ) * inv;
    }
    G[idx] = (bf16_t)v;
  }
}

// res = qf * conj(kf) * weight, on transposed freq layout [d][f] (stride FPAD)
__global__ void cmul(const bf16_t* __restrict__ QF, const bf16_t* __restrict__ KF,
                     const float* __restrict__ W, bf16_t* __restrict__ RES)
{
  const long long idx = (long long)blockIdx.x * blockDim.x + threadIdx.x;
  if (idx >= (long long)BHN * DKD * NFREQ) return;
  const int f  = (int)(idx % NFREQ);
  const long long r = idx / NFREQ;
  const int d  = (int)(r % DKD);
  const int bh = (int)(r / DKD);
  const int h  = bh & (NH - 1);
  const size_t base = (size_t)bh * DKD * FPAD + (size_t)d * FPAD;
  const float qr = (float)QF[base + f];
  const float qi = (float)QF[base + NFREQ + f];
  const float kr = (float)KF[base + f];
  const float ki = (float)KF[base + NFREQ + f];
  const size_t wix = (((size_t)h * NFREQ + f) * DKD + d) * 2;
  const float wr = W[wix], wi = W[wix + 1];
  const float ar = qr * kr + qi * ki;      // Re(q * conj(k))
  const float ai = qi * kr - qr * ki;      // Im(q * conj(k))
  RES[base + f]         = (bf16_t)(ar * wr - ai * wi);
  RES[base + NFREQ + f] = (bf16_t)(ar * wi + ai * wr);
}

// row-wise softmax (rows are contiguous, length SEQ); one wave per row
__global__ __launch_bounds__(256) void softmax_rows(const float* __restrict__ A,
                                                    bf16_t* __restrict__ O)
{
  const int lane = threadIdx.x & 31;
  const int row  = blockIdx.x * (blockDim.x >> 5) + (threadIdx.x >> 5);
  if (row >= BHN * SEQ) return;
  const float* a = A + (size_t)row * SEQ;
  bf16_t* o = O + (size_t)row * SEQ;
  float mx = -1e30f;
  for (int t = lane; t < SEQ; t += 32) mx = fmaxf(mx, a[t]);
  for (int s = 16; s > 0; s >>= 1) mx = fmaxf(mx, __shfl_xor(mx, s, 32));
  float sum = 0.0f;
  for (int t = lane; t < SEQ; t += 32) sum += __expf(a[t] - mx);
  for (int s = 16; s > 0; s >>= 1) sum += __shfl_xor(sum, s, 32);
  const float inv = 1.0f / sum;
  for (int t = lane; t < SEQ; t += 32) o[t] = (bf16_t)(__expf(a[t] - mx) * inv);
}

// fused GeLU + LayerNorm over rows of HIDN
__global__ __launch_bounds__(256) void gelu_ln(
    const float* __restrict__ X, const float* __restrict__ lnw,
    const float* __restrict__ lnb, float* __restrict__ out)
{
  __shared__ float red[256];
  const int row = blockIdx.x;
  const int tid = threadIdx.x;
  const float* x = X + (size_t)row * HIDN;
  float* o = out + (size_t)row * HIDN;
  float g[HIDN / 256];
  float lsum = 0.0f;
#pragma unroll
  for (int i = 0; i < HIDN / 256; ++i) {
    const float v = x[tid + 256 * i];
    const float t = v * 0.5f * (1.0f + erff(v * 0.70710678f));
    g[i] = t; lsum += t;
  }
  red[tid] = lsum; __syncthreads();
  for (int s = 128; s > 0; s >>= 1) { if (tid < s) red[tid] += red[tid + s]; __syncthreads(); }
  const float mean = red[0] * (1.0f / HIDN);
  __syncthreads();
  float lvar = 0.0f;
#pragma unroll
  for (int i = 0; i < HIDN / 256; ++i) { const float d = g[i] - mean; lvar += d * d; }
  red[tid] = lvar; __syncthreads();
  for (int s = 128; s > 0; s >>= 1) { if (tid < s) red[tid] += red[tid + s]; __syncthreads(); }
  const float inv = rsqrtf(red[0] * (1.0f / HIDN) + 1e-12f);
#pragma unroll
  for (int i = 0; i < HIDN / 256; ++i) {
    const int c = tid + 256 * i;
    o[c] = lnw[c] * ((g[i] - mean) * inv) + lnb[c];
  }
}

// ---------------------------------------------------------------------------
extern "C" void kernel_launch(void* const* d_in, const int* in_sizes, int n_in,
                              void* d_out, int out_size, void* d_ws, size_t ws_size,
                              hipStream_t stream)
{
  (void)in_sizes; (void)n_in; (void)out_size; (void)ws_size;
  const float* query = (const float*)d_in[0];
  const float* key   = (const float*)d_in[1];
  const float* value = (const float*)d_in[2];
  const float* w0    = (const float*)d_in[3];
  const float* w1    = (const float*)d_in[4];
  const float* w2    = (const float*)d_in[5];
  const float* w3    = (const float*)d_in[6];
  const float* abias = (const float*)d_in[7];
  const float* cwght = (const float*)d_in[8];
  const float* linw  = (const float*)d_in[9];
  const float* linb  = (const float*)d_in[10];
  const float* lnw   = (const float*)d_in[11];
  const float* lnb   = (const float*)d_in[12];
  float* out = (float*)d_out;

  char* ws = (char*)d_ws;
  size_t off = 0;
  auto take = [&](size_t bytes) -> size_t {
    size_t o = off; off += (bytes + 255) & ~(size_t)255; return o;
  };
  const size_t nQKV = (size_t)NBATCH * SEQ * HIDN;     // 26.2M
  const size_t sPT  = (size_t)SEQ * DKD;               // 51200  per (b,h)
  const size_t sFQ  = (size_t)DKD * FPAD;              // 53248  per (b,h)
  const size_t sAl  = (size_t)SEQ * SEQ;               // 40000  per (b,h)

  bf16_t* Fb   = (bf16_t*)(ws + take((size_t)FROWS * SEQ * 2));
  bf16_t* Gb   = (bf16_t*)(ws + take((size_t)SEQ * FROWS * 2));
  bf16_t* w0T  = (bf16_t*)(ws + take((size_t)DKD * DKD * 2));
  bf16_t* w1T  = (bf16_t*)(ws + take((size_t)DKD * DKD * 2));
  bf16_t* w2T  = (bf16_t*)(ws + take((size_t)DKD * DKD * 2));
  bf16_t* w3b  = (bf16_t*)(ws + take((size_t)SEQ * DKD * 2));
  bf16_t* linb16 = (bf16_t*)(ws + take((size_t)HIDN * HIDN * 2));
  bf16_t* qb   = (bf16_t*)(ws + take(nQKV * 2));
  bf16_t* kb   = (bf16_t*)(ws + take(nQKV * 2));
  bf16_t* vb   = (bf16_t*)(ws + take(nQKV * 2));
  bf16_t* qpT  = (bf16_t*)(ws + take((size_t)BHN * sPT * 2));
  bf16_t* kpT  = (bf16_t*)(ws + take((size_t)BHN * sPT * 2));
  bf16_t* vpT  = (bf16_t*)(ws + take((size_t)BHN * sPT * 2));
  bf16_t* QFT  = (bf16_t*)(ws + take((size_t)BHN * sFQ * 2));
  bf16_t* KFT  = (bf16_t*)(ws + take((size_t)BHN * sFQ * 2));
  bf16_t* REST = (bf16_t*)(ws + take((size_t)BHN * sFQ * 2));
  bf16_t* corr = (bf16_t*)(ws + take((size_t)BHN * sPT * 2));
  bf16_t* alpB = (bf16_t*)(ws + take((size_t)BHN * sAl * 2));
  bf16_t* xm   = (bf16_t*)(ws + take(nQKV * 2));
  // region reuse (safe by sequencing):
  float* alpha = (float*)qb;    // 82MB <= qb+kb (105MB); qb/kb dead after projections
  float* x2    = (float*)QFT;   // 105MB <= QFT+KFT (109MB); dead after cmul

  // ---- 0) one-time converts + DFT matrices ----
  cvt_f32_bf16<<<(unsigned)((nQKV + 255) / 256), 256, 0, stream>>>(query, qb, (long long)nQKV);
  cvt_f32_bf16<<<(unsigned)((nQKV + 255) / 256), 256, 0, stream>>>(key,   kb, (long long)nQKV);
  cvt_f32_bf16<<<(unsigned)((nQKV + 255) / 256), 256, 0, stream>>>(value, vb, (long long)nQKV);
  cvt_f32_bf16_t<<<(DKD * DKD + 255) / 256, 256, 0, stream>>>(w0, w0T, DKD, DKD);
  cvt_f32_bf16_t<<<(DKD * DKD + 255) / 256, 256, 0, stream>>>(w1, w1T, DKD, DKD);
  cvt_f32_bf16_t<<<(DKD * DKD + 255) / 256, 256, 0, stream>>>(w2, w2T, DKD, DKD);
  cvt_f32_bf16<<<(SEQ * DKD + 255) / 256, 256, 0, stream>>>(w3, w3b, SEQ * DKD);
  cvt_f32_bf16<<<(HIDN * HIDN + 255) / 256, 256, 0, stream>>>(linw, linb16, HIDN * HIDN);
  init_dft<<<(FROWS * SEQ + 255) / 256, 256, 0, stream>>>(Fb, Gb);

  const dim3 blk(NTHR);
  // ---- 1) projections -> qpT/kpT/vpT[d][s]  (transposed bf16 out) ----
  {
    const dim3 grid((DKD + BN - 1) / BN, (SEQ + BM - 1) / BM, BHN);
    wmma_gemm<EPI_T_BF16><<<grid, blk, 0, stream>>>(
        qb, HIDN, (long long)SEQ * HIDN, DKD,  w0T, DKD, 0, 0,
        qpT, SEQ, (long long)NH * sPT, (long long)sPT, nullptr, SEQ, DKD, DKD, NH);
    wmma_gemm<EPI_T_BF16><<<grid, blk, 0, stream>>>(
        kb, HIDN, (long long)SEQ * HIDN, DKD,  w1T, DKD, 0, 0,
        kpT, SEQ, (long long)NH * sPT, (long long)sPT, nullptr, SEQ, DKD, DKD, NH);
    wmma_gemm<EPI_T_BF16><<<grid, blk, 0, stream>>>(
        vb, HIDN, (long long)SEQ * HIDN, DKD,  w2T, DKD, 0, 0,
        vpT, SEQ, (long long)NH * sPT, (long long)sPT, nullptr, SEQ, DKD, DKD, NH);
  }
  // ---- 2) rfft: [Re;Im] = F @ qp -> QFT/KFT[d][f] (transposed, stride FPAD) ----
  {
    const dim3 grid((DKD + BN - 1) / BN, (FROWS + BM - 1) / BM, BHN);
    wmma_gemm<EPI_T_BF16><<<grid, blk, 0, stream>>>(
        Fb, SEQ, 0, 0,  qpT, SEQ, (long long)NH * sPT, (long long)sPT,
        QFT, FPAD, (long long)NH * sFQ, (long long)sFQ, nullptr, FROWS, DKD, SEQ, NH);
    wmma_gemm<EPI_T_BF16><<<grid, blk, 0, stream>>>(
        Fb, SEQ, 0, 0,  kpT, SEQ, (long long)NH * sPT, (long long)sPT,
        KFT, FPAD, (long long)NH * sFQ, (long long)sFQ, nullptr, FROWS, DKD, SEQ, NH);
  }
  // ---- 3) res = qf * conj(kf) * weight ----
  {
    const long long tot = (long long)BHN * DKD * NFREQ;
    cmul<<<(unsigned)((tot + 255) / 256), 256, 0, stream>>>(QFT, KFT, cwght, REST);
  }
  // ---- 4) irfft + bias + relu -> corr[t][d] (normal bf16) ----
  {
    const dim3 grid((DKD + BN - 1) / BN, (SEQ + BM - 1) / BM, BHN);
    wmma_gemm<EPI_BIAS_RELU_BF16><<<grid, blk, 0, stream>>>(
        Gb, FROWS, 0, 0,  REST, FPAD, (long long)NH * sFQ, (long long)sFQ,
        corr, DKD, (long long)NH * sPT, (long long)sPT, abias, SEQ, DKD, FROWS, NH);
  }
  // ---- 5) alphaT[l][t] = w3 @ corr^T  (f32, rows contiguous in t) ----
  {
    const dim3 grid((SEQ + BN - 1) / BN, (SEQ + BM - 1) / BM, BHN);
    wmma_gemm<EPI_F32><<<grid, blk, 0, stream>>>(
        w3b, DKD, 0, 0,  corr, DKD, (long long)NH * sPT, (long long)sPT,
        alpha, SEQ, (long long)NH * sAl, (long long)sAl, nullptr, SEQ, SEQ, DKD, NH);
  }
  // ---- 6) softmax over t (contiguous rows) -> bf16 ----
  softmax_rows<<<(BHN * SEQ + 7) / 8, 256, 0, stream>>>(alpha, alpB);

  // ---- 7) x[l][d] = alphaT @ vp, head-merged into xm (B*SEQ, HIDN) bf16 ----
  {
    const dim3 grid((DKD + BN - 1) / BN, (SEQ + BM - 1) / BM, BHN);
    wmma_gemm<EPI_BF16><<<grid, blk, 0, stream>>>(
        alpB, SEQ, (long long)NH * sAl, (long long)sAl,
        vpT, SEQ, (long long)NH * sPT, (long long)sPT,
        xm, HIDN, (long long)SEQ * HIDN, DKD, nullptr, SEQ, DKD, SEQ, NH);
  }
  // ---- 8) x2 = xm @ linear_w^T + linear_b (f32) ----
  {
    const dim3 grid((HIDN + BN - 1) / BN, (NBATCH * SEQ + BM - 1) / BM, 1);
    wmma_gemm<EPI_BIAS_F32><<<grid, blk, 0, stream>>>(
        xm, HIDN, 0, 0,  linb16, HIDN, 0, 0,
        x2, HIDN, 0, 0, linb, NBATCH * SEQ, HIDN, HIDN, 1);
  }
  // ---- 9) GeLU + LayerNorm -> out ----
  gelu_ln<<<NBATCH * SEQ, 256, 0, stream>>>(x2, lnw, lnb, out);
}
